// CBLLoss_39444979647062
// MI455X (gfx1250) — compile-verified, matrix operand order checked by reading, not run on hardware
//
#include <hip/hip_runtime.h>
#include <hip/hip_bf16.h>
#include <math.h>

// CDNA5 / gfx1250, wave32.
typedef __attribute__((ext_vector_type(16))) _Float16 v16h;
typedef __attribute__((ext_vector_type(8)))  float    v8f;

#define EPS_F 1e-12f
#define C_DIM 128          // feature dim (reference)
#define K_NBR 7            // NSAMPLE-1 neighbors (reference)
#define WAVES_PER_BLOCK 8  // 256 threads = 8 wave32s

__global__ void cbl_init_kernel(float* accum) {
    accum[0] = 0.0f;   // sum of selected per-pair losses
    accum[1] = 0.0f;   // count of selected pairs
}

// One wave per point. Rows 0..7 of the WMMA A-tile = [self, nbr0..nbr6],
// rows 8..15 zero. G = A*A^T via chained v_wmma_f32_16x16x32_f16 over four
// K=32 chunks: G[0][j] = dot(x_i, x_row_j) (lanes 0..15, acc[0]),
// G[j][j] = |x_row_j|^2 (lane j, acc[j]) -> normalization for free.
__global__ __launch_bounds__(256) void cbl_main_kernel(
    const float* __restrict__ x,        // [N, 128] f32
    const int*   __restrict__ nidx,     // [N, 7]
    const int*   __restrict__ target,   // [N]
    float*       __restrict__ accum,    // {sum, cnt}
    int N)
{
    __shared__ float sSum[WAVES_PER_BLOCK];
    __shared__ float sCnt[WAVES_PER_BLOCK];

    const int lane  = threadIdx.x & 31;
    const int wid   = threadIdx.x >> 5;
    const int point = blockIdx.x * WAVES_PER_BLOCK + wid;

    if (lane == 0) { sSum[wid] = 0.0f; sCnt[wid] = 0.0f; }

    if (point < N) {  // wave-uniform: EXEC all-ones inside (WMMA requirement)
        const int M  = lane & 15;   // tile row this lane feeds
        const int kh = lane >> 4;   // K half: lanes 0-15 -> K 0..15, 16-31 -> K 16..31

        int row = point;
        if (M >= 1 && M <= K_NBR)
            row = nidx[(size_t)point * K_NBR + (M - 1)];
        const bool active = (M <= K_NBR);

        const int   t    = active ? target[row] : 0;
        const float* xrow = x + (size_t)row * C_DIM;

        v8f acc = {};
        #pragma unroll
        for (int ch = 0; ch < 4; ++ch) {          // K chunks of 32
            v16h a = {};
            if (active) {
                const float4* rp =
                    reinterpret_cast<const float4*>(xrow + ch * 32 + kh * 16);
                float4 f0 = rp[0], f1 = rp[1], f2 = rp[2], f3 = rp[3];
                a[0]=(_Float16)f0.x;  a[1]=(_Float16)f0.y;  a[2]=(_Float16)f0.z;  a[3]=(_Float16)f0.w;
                a[4]=(_Float16)f1.x;  a[5]=(_Float16)f1.y;  a[6]=(_Float16)f1.z;  a[7]=(_Float16)f1.w;
                a[8]=(_Float16)f2.x;  a[9]=(_Float16)f2.y;  a[10]=(_Float16)f2.z; a[11]=(_Float16)f2.w;
                a[12]=(_Float16)f3.x; a[13]=(_Float16)f3.y; a[14]=(_Float16)f3.z; a[15]=(_Float16)f3.w;
            }
            // Gram accumulate: same fragment as A and B (B column layout
            // mirrors A row layout) -> acc += A_chunk * A_chunk^T
            acc = __builtin_amdgcn_wmma_f32_16x16x32_f16(
                false, a, false, a, (short)0, acc, false, false);
        }

        // Diagonal s = G[row,row]: lives at lane==row, acc[row] (rows 0..7).
        float s = acc[0];
        #pragma unroll
        for (int q = 1; q < 8; ++q)
            if ((lane & 7) == q) s = acc[q];

        const float dot0 = acc[0];                 // lanes 0..15: G[0][lane]
        const float invn = rsqrtf(s + EPS_F);      // 1/sqrt(|x|^2+eps)
        const float na   = s * invn * invn;        // |feats|^2 = s/(s+eps)

        const float invn0 = __shfl(invn, 0);
        const float na0   = __shfl(na,   0);
        const int   t0    = __shfl(t,    0);

        const float d2   = na0 + na - 2.0f * dot0 * invn0 * invn;
        const float dist = sqrtf(fmaxf(d2, 0.0f) + EPS_F);
        const int   pos  = (t == t0) ? 1 : 0;

        // Gather the 7 neighbor results to every lane (shuffles need full EXEC)
        float dj[K_NBR]; int pj[K_NBR];
        #pragma unroll
        for (int j = 1; j <= K_NBR; ++j) {
            dj[j - 1] = __shfl(dist, j);
            pj[j - 1] = __shfl(pos,  j);
        }

        if (lane == 0) {
            int npos = 0;
            #pragma unroll
            for (int j = 0; j < K_NBR; ++j) npos += pj[j];
            if (npos > 0 && npos < K_NBR) {        // point_mask
                float dmax = -dj[0];
                #pragma unroll
                for (int j = 1; j < K_NBR; ++j) dmax = fmaxf(dmax, -dj[j]);
                float e[K_NBR], neg = 0.0f;
                #pragma unroll
                for (int j = 0; j < K_NBR; ++j) {
                    e[j] = expf(-dj[j] - dmax);
                    neg += e[j] * (1.0f - (float)pj[j]);
                }
                float ls = 0.0f;
                #pragma unroll
                for (int j = 0; j < K_NBR; ++j)
                    if (pj[j]) ls += logf(e[j] + neg) - (-dj[j] - dmax);
                sSum[wid] = ls;
                sCnt[wid] = (float)npos;
            }
        }
    }

    __syncthreads();
    if (threadIdx.x == 0) {
        float bs = 0.0f, bc = 0.0f;
        #pragma unroll
        for (int w = 0; w < WAVES_PER_BLOCK; ++w) { bs += sSum[w]; bc += sCnt[w]; }
        if (bc > 0.0f) {
            atomicAdd(&accum[0], bs);
            atomicAdd(&accum[1], bc);
        }
    }
}

__global__ void cbl_finalize_kernel(const float* accum, float* out) {
    const float s = accum[0], c = accum[1];
    out[0] = (c > 0.0f) ? (s / fmaxf(c, 1.0f)) : 0.0f;
}

extern "C" void kernel_launch(void* const* d_in, const int* in_sizes, int n_in,
                              void* d_out, int out_size, void* d_ws, size_t ws_size,
                              hipStream_t stream) {
    // inputs: 0=p (unused), 1=x [N,128] f32, 2=neighbor_idx [N,7] int, 3=target [N] int
    const float* x      = (const float*)d_in[1];
    const int*   nidx   = (const int*)  d_in[2];
    const int*   target = (const int*)  d_in[3];
    float*       out    = (float*)d_out;
    float*       accum  = (float*)d_ws;          // {sum, cnt}

    const int N = in_sizes[3];

    cbl_init_kernel<<<1, 1, 0, stream>>>(accum);
    const int blocks = (N + WAVES_PER_BLOCK - 1) / WAVES_PER_BLOCK;
    cbl_main_kernel<<<blocks, 256, 0, stream>>>(x, nidx, target, accum, N);
    cbl_finalize_kernel<<<1, 1, 0, stream>>>(accum, out);
}
